// AttentionBiasHead_65077344469128
// MI455X (gfx1250) — matrix operand order repaired and checked by reading, not compiled, not run on hardware
//
#include <hip/hip_runtime.h>
#include <cstdint>

// Problem constants (match reference)
#define B_   16
#define L_   2048
#define DIN_ 1024
#define DQ_  128
#define HID_ 128

typedef __attribute__((ext_vector_type(16))) __bf16 v16bf;
typedef __attribute__((ext_vector_type(8)))  __bf16 v8bf;
typedef __attribute__((ext_vector_type(8)))  float  v8f;

// ---------------------------------------------------------------- WMMA core
static __device__ __forceinline__ v8f wmma_bf16(v16bf a, v16bf b, v8f c) {
  return __builtin_amdgcn_wmma_f32_16x16x32_bf16(false, a, false, b, (short)0, c,
                                                 false, false);
}

static __device__ __forceinline__ v16bf cat8(v8bf lo, v8bf hi) {
  return __builtin_shufflevector(lo, hi, 0,1,2,3,4,5,6,7,8,9,10,11,12,13,14,15);
}

// A fragment (16 rows x 32 K, bf16). Per ISA: lane holds row M=lane%16 and two
// contiguous 8-element K runs at K = 8*(lane/16) and 16 + 8*(lane/16).
static __device__ __forceinline__ v16bf frag_a(const __bf16* tile, int lane, int stride) {
  int row = lane & 15, kg = (lane >> 4) & 1;
  const __bf16* p = tile + row * stride + kg * 8;
  return cat8(*(const v8bf*)p, *(const v8bf*)(p + 16));
}

// B fragment (32 K x 16 N, bf16). Lane holds column N=lane%16, K = 8*(lane/16)+s and 16+8*(lane/16)+s.
static __device__ __forceinline__ v16bf frag_b(const __bf16* tile, int lane, int stride, int n0) {
  int kg = (lane >> 4) & 1;
  const __bf16* p = tile + n0 + (lane & 15);
  v16bf v;
#pragma unroll
  for (int s = 0; s < 8; ++s) {
    v[s] = p[(kg * 8 + s) * stride];
    v[8 + s] = p[(16 + kg * 8 + s) * stride];
  }
  return v;
}

// -------------------------------------------------- async LDS staging (CDNA5)
// 64KB bf16 block (4096 x 16B chunks) staged via GLOBAL_LOAD_ASYNC_TO_LDS_B128 (GVS mode:
// saddr = uniform base, vaddr = 32-bit byte offset, vdst = LDS byte offset).
static __device__ __forceinline__ void async_tile_copy(__bf16* dst, const __bf16* src, int tid) {
  unsigned long long gb = (unsigned long long)(uintptr_t)src;
  unsigned lb = (unsigned)(uintptr_t)dst;  // addr[31:0] of LDS flat addr == LDS offset
#pragma unroll
  for (int i = 0; i < 16; ++i) {
    int idx = tid + i * 256;               // 4096 chunks of 16 bytes
    unsigned loff = lb + (unsigned)(idx * 16);
    unsigned goff = (unsigned)(idx * 16);
    asm volatile("global_load_async_to_lds_b128 %0, %1, %2"
                 :: "v"(loff), "v"(goff), "s"(gb) : "memory");
  }
}
static __device__ __forceinline__ void async_wait() {
  asm volatile("s_wait_asynccnt 0" ::: "memory");
}

// Stage a 128x128 bf16 tile transposed into LDS: dstT[d][j] = src[j][d]
static __device__ __forceinline__ void stage_transpose(__bf16* dstT, const __bf16* src, int tid) {
#pragma unroll
  for (int i = 0; i < 8; ++i) {
    int idx = tid + i * 256;               // 2048 chunks of 8 elements
    int j  = idx >> 4;
    int d8 = (idx & 15) * 8;
    v8bf v = *(const v8bf*)(src + j * 128 + d8);
#pragma unroll
    for (int e = 0; e < 8; ++e) dstT[(d8 + e) * 128 + j] = v[e];
  }
}

static __device__ __forceinline__ float hmax16(float v) {
  v = fmaxf(v, __shfl_xor(v, 1, 32));
  v = fmaxf(v, __shfl_xor(v, 2, 32));
  v = fmaxf(v, __shfl_xor(v, 4, 32));
  v = fmaxf(v, __shfl_xor(v, 8, 32));
  return v;
}
static __device__ __forceinline__ float hsum16(float v) {
  v += __shfl_xor(v, 1, 32);
  v += __shfl_xor(v, 2, 32);
  v += __shfl_xor(v, 4, 32);
  v += __shfl_xor(v, 8, 32);
  return v;
}

// ------------------------------------------------------------- small kernels
__global__ void kcvt(const float* __restrict__ s, __bf16* __restrict__ d, int n) {
  int i = blockIdx.x * blockDim.x + threadIdx.x;
  if (i < n) d[i] = (__bf16)s[i];
}

__global__ void knzero(float* p) { if (threadIdx.x == 0) *p = 0.0f; }

// Asf[b][j] = b1[j] + sum_d sf[b,d]*W1[d][j];  Ppos[l][j] = sum_d pos[l,d]*W1[64+d][j]
__global__ void kprep(const float* __restrict__ sf, const float* __restrict__ pos,
                      const float* __restrict__ W1, const float* __restrict__ b1,
                      float* __restrict__ asf, float* __restrict__ ppos) {
  int idx = blockIdx.x * blockDim.x + threadIdx.x;
  const int total = (B_ + L_) * HID_;
  if (idx >= total) return;
  if (idx < B_ * HID_) {
    int b = idx >> 7, j = idx & 127;
    float s = b1[j];
    for (int d = 0; d < 64; ++d) s += sf[b * 64 + d] * W1[d * HID_ + j];
    asf[idx] = s;
  } else {
    int t = idx - B_ * HID_;
    int l = t >> 7, j = t & 127;
    float s = 0.0f;
    for (int d = 0; d < 32; ++d) s += pos[l * 32 + d] * W1[(64 + d) * HID_ + j];
    ppos[t] = s;
  }
}

// ------------------------------------------------------------ generic GEMM
// C[128 rows x 128 cols] = act_tile @ W + bias, A fp32 (or CPE on-the-fly),
// W bf16 [K][128], out bf16 or f32. 256 threads = 8 waves; wave = 2x4 WMMA tiles.
__global__ __launch_bounds__(256, 1)
void kgemm(const float* __restrict__ A, const __bf16* __restrict__ W,
           const float* __restrict__ bias, void* __restrict__ outp,
           int K, int lda, int mode, int outBf,
           const float* __restrict__ asf, const float* __restrict__ ppos) {
  __shared__ __bf16 sA[128 * 32];
  __shared__ __bf16 sB[32 * 128];
  int tid = threadIdx.x, lane = tid & 31, wave = tid >> 5;
  int rb = wave >> 1, cg = wave & 1;
  int hf = lane >> 4, ln = lane & 15;
  int row0 = blockIdx.x * 128;

  v8f acc[2][4];
#pragma unroll
  for (int t = 0; t < 2; ++t)
#pragma unroll
    for (int c = 0; c < 4; ++c) acc[t][c] = (v8f){0,0,0,0,0,0,0,0};

  for (int k0 = 0; k0 < K; k0 += 32) {
    // Stage A tile 128x32 fp32 -> bf16
#pragma unroll
    for (int i = 0; i < 4; ++i) {
      int idx = tid + i * 256;            // 1024 chunks of 4 floats
      int r = idx >> 3, c4 = (idx & 7) * 4;
      int grow = row0 + r;
      float v0, v1, v2, v3;
      if (mode == 0) {
        const float4 f = *(const float4*)(A + (size_t)grow * lda + k0 + c4);
        v0 = f.x; v1 = f.y; v2 = f.z; v3 = f.w;
      } else {
        int bb = grow >> 11, ll = grow & 2047;
        const float* pa = asf + bb * HID_ + k0 + c4;
        const float* pp = ppos + (size_t)ll * HID_ + k0 + c4;
        v0 = fmaxf(pa[0] + pp[0], 0.0f);
        v1 = fmaxf(pa[1] + pp[1], 0.0f);
        v2 = fmaxf(pa[2] + pp[2], 0.0f);
        v3 = fmaxf(pa[3] + pp[3], 0.0f);
      }
      __bf16* d = sA + r * 32 + c4;
      d[0] = (__bf16)v0; d[1] = (__bf16)v1; d[2] = (__bf16)v2; d[3] = (__bf16)v3;
    }
    // Stage B tile 32x128 (bf16 weights already converted)
#pragma unroll
    for (int i = 0; i < 2; ++i) {
      int idx = tid + i * 256;            // 512 chunks of 8
      int kr = idx >> 4, c8 = (idx & 15) * 8;
      *(v8bf*)(sB + kr * 128 + c8) = *(const v8bf*)(W + (size_t)(k0 + kr) * 128 + c8);
    }
    __syncthreads();

    v16bf a0 = frag_a(sA + (rb * 32 + 0) * 32, lane, 32);
    v16bf a1 = frag_a(sA + (rb * 32 + 16) * 32, lane, 32);
#pragma unroll
    for (int c = 0; c < 4; ++c) {
      v16bf bfr = frag_b(sB, lane, 128, cg * 64 + c * 16);
      acc[0][c] = wmma_bf16(a0, bfr, acc[0][c]);
      acc[1][c] = wmma_bf16(a1, bfr, acc[1][c]);
    }
    __syncthreads();
  }

  // Epilogue: C layout row = r + 8*(lane/16), col = lane%16 within 16x16 tile
#pragma unroll
  for (int t = 0; t < 2; ++t)
#pragma unroll
    for (int c = 0; c < 4; ++c)
#pragma unroll
      for (int r = 0; r < 8; ++r) {
        int row = row0 + rb * 32 + t * 16 + hf * 8 + r;
        int col = cg * 64 + c * 16 + ln;
        float v = acc[t][c][r] + bias[col];
        if (outBf) ((__bf16*)outp)[(size_t)row * 128 + col] = (__bf16)v;
        else       ((float*)outp)[(size_t)row * 128 + col] = v;
      }
}

// --------------------------------------------------------- flash attention
// out[b, i0:i0+128, :] = softmax(q k^T / sqrt(dq)) @ v   (online softmax)
__global__ __launch_bounds__(256, 1)
void kflash(const __bf16* __restrict__ qb, const __bf16* __restrict__ kb,
            const __bf16* __restrict__ vb, float* __restrict__ out) {
  __shared__ __bf16 sQ[128 * 128];
  __shared__ __bf16 sKT[128 * 128];
  __shared__ __bf16 sV[128 * 128];
  __shared__ __bf16 sP[128 * 128];
  __shared__ float sRedM[2][128];
  __shared__ float sRedS[2][128];

  int tid = threadIdx.x, lane = tid & 31, wave = tid >> 5;
  int rb = wave >> 1, cg = wave & 1, hf = lane >> 4, ln = lane & 15;
  int b = blockIdx.y, i0 = blockIdx.x * 128;
  const float scale = 0.08838834764831845f;  // 1/sqrt(128)

  async_tile_copy(sQ, qb + ((size_t)b * L_ + i0) * 128, tid);

  v8f acc[2][4];
  float ms[16], lsum[16];
#pragma unroll
  for (int t = 0; t < 2; ++t)
#pragma unroll
    for (int c = 0; c < 4; ++c) acc[t][c] = (v8f){0,0,0,0,0,0,0,0};
#pragma unroll
  for (int i = 0; i < 16; ++i) { ms[i] = -3.0e38f; lsum[i] = 0.0f; }

  for (int j0 = 0; j0 < L_; j0 += 128) {
    stage_transpose(sKT, kb + ((size_t)b * L_ + j0) * 128, tid);
    async_tile_copy(sV, vb + ((size_t)b * L_ + j0) * 128, tid);
    async_wait();
    __syncthreads();

    // S = q k^T (K dim = 128 over dq)
    v8f ps[2][4];
#pragma unroll
    for (int t = 0; t < 2; ++t)
#pragma unroll
      for (int c = 0; c < 4; ++c) ps[t][c] = (v8f){0,0,0,0,0,0,0,0};
#pragma unroll
    for (int kk = 0; kk < 4; ++kk) {
      v16bf a0 = frag_a(sQ + (rb * 32 + 0) * 128 + kk * 32, lane, 128);
      v16bf a1 = frag_a(sQ + (rb * 32 + 16) * 128 + kk * 32, lane, 128);
#pragma unroll
      for (int c = 0; c < 4; ++c) {
        v16bf bfr = frag_b(sKT + (kk * 32) * 128, lane, 128, cg * 64 + c * 16);
        ps[0][c] = wmma_bf16(a0, bfr, ps[0][c]);
        ps[1][c] = wmma_bf16(a1, bfr, ps[1][c]);
      }
    }

    // scale + per-row max (across this wave's 64 cols)
    float rm[16];
#pragma unroll
    for (int t = 0; t < 2; ++t)
#pragma unroll
      for (int r = 0; r < 8; ++r) {
        float m = -3.0e38f;
#pragma unroll
        for (int c = 0; c < 4; ++c) {
          ps[t][c][r] *= scale;
          m = fmaxf(m, ps[t][c][r]);
        }
        rm[t * 8 + r] = hmax16(m);
      }
    if (ln == 0) {
#pragma unroll
      for (int t = 0; t < 2; ++t)
#pragma unroll
        for (int r = 0; r < 8; ++r)
          sRedM[cg][rb * 32 + t * 16 + hf * 8 + r] = rm[t * 8 + r];
    }
    __syncthreads();

    // combine maxes across the two col-groups, rescale acc/l, exponentiate
    float rs[16];
#pragma unroll
    for (int t = 0; t < 2; ++t)
#pragma unroll
      for (int r = 0; r < 8; ++r) {
        int row = rb * 32 + t * 16 + hf * 8 + r;
        float mn = fmaxf(rm[t * 8 + r], sRedM[cg ^ 1][row]);
        float sc = __expf(ms[t * 8 + r] - mn);
        ms[t * 8 + r] = mn;
        lsum[t * 8 + r] *= sc;
#pragma unroll
        for (int c = 0; c < 4; ++c) acc[t][c][r] *= sc;
        float s = 0.0f;
#pragma unroll
        for (int c = 0; c < 4; ++c) {
          float p = __expf(ps[t][c][r] - mn);
          s += p;
          sP[row * 128 + cg * 64 + c * 16 + ln] = (__bf16)p;
        }
        rs[t * 8 + r] = hsum16(s);
      }
    if (ln == 0) {
#pragma unroll
      for (int t = 0; t < 2; ++t)
#pragma unroll
        for (int r = 0; r < 8; ++r)
          sRedS[cg][rb * 32 + t * 16 + hf * 8 + r] = rs[t * 8 + r];
    }
    __syncthreads();
#pragma unroll
    for (int t = 0; t < 2; ++t)
#pragma unroll
      for (int r = 0; r < 8; ++r) {
        int row = rb * 32 + t * 16 + hf * 8 + r;
        lsum[t * 8 + r] += rs[t * 8 + r] + sRedS[cg ^ 1][row];
      }

    // acc += P @ V  (K dim = 128 over j)
#pragma unroll
    for (int kk = 0; kk < 4; ++kk) {
      v16bf a0 = frag_a(sP + (rb * 32 + 0) * 128 + kk * 32, lane, 128);
      v16bf a1 = frag_a(sP + (rb * 32 + 16) * 128 + kk * 32, lane, 128);
#pragma unroll
      for (int c = 0; c < 4; ++c) {
        v16bf bfr = frag_b(sV + (kk * 32) * 128, lane, 128, cg * 64 + c * 16);
        acc[0][c] = wmma_bf16(a0, bfr, acc[0][c]);
        acc[1][c] = wmma_bf16(a1, bfr, acc[1][c]);
      }
    }
    __syncthreads();
  }

#pragma unroll
  for (int t = 0; t < 2; ++t)
#pragma unroll
    for (int c = 0; c < 4; ++c)
#pragma unroll
      for (int r = 0; r < 8; ++r) {
        int row = i0 + rb * 32 + t * 16 + hf * 8 + r;
        int col = cg * 64 + c * 16 + ln;
        out[((size_t)b * L_ + row) * DQ_ + col] = acc[t][c][r] / lsum[t * 8 + r];
      }
}

// ------------------------------------------------- bias @ V + L1 diff-norm
__global__ __launch_bounds__(256, 1)
void kbias(const __bf16* __restrict__ qcb, const __bf16* __restrict__ kcb,
           const __bf16* __restrict__ vb, float* __restrict__ out,
           float* __restrict__ normp) {
  __shared__ __bf16 sQC[128 * 128];
  __shared__ __bf16 sKCT[128 * 128];
  __shared__ __bf16 sV[128 * 128];
  __shared__ __bf16 sP[128 * 128];   // bias tile (bf16)
  __shared__ float  sPrevCol[128];   // bias[:, j0-1]
  __shared__ __bf16 sQm1[128];       // qc row i0-1
  __shared__ float  sProw[128];      // bias[i0-1, j-tile]

  int tid = threadIdx.x, lane = tid & 31, wave = tid >> 5;
  int rb = wave >> 1, cg = wave & 1, hf = lane >> 4, ln = lane & 15;
  int b = blockIdx.y, i0 = blockIdx.x * 128;

  async_tile_copy(sQC, qcb + ((size_t)b * L_ + i0) * 128, tid);
  if (i0 > 0 && tid < 16)
    *(v8bf*)(sQm1 + tid * 8) =
        *(const v8bf*)(qcb + ((size_t)b * L_ + i0 - 1) * 128 + tid * 8);

  v8f accb[2][4];
#pragma unroll
  for (int t = 0; t < 2; ++t)
#pragma unroll
    for (int c = 0; c < 4; ++c) accb[t][c] = (v8f){0,0,0,0,0,0,0,0};
  float norm = 0.0f;

  for (int j0 = 0; j0 < L_; j0 += 128) {
    stage_transpose(sKCT, kcb + ((size_t)b * L_ + j0) * 128, tid);
    async_tile_copy(sV, vb + ((size_t)b * L_ + j0) * 128, tid);
    async_wait();
    __syncthreads();

    // bias tile = qc kc^T
    v8f bs[2][4];
#pragma unroll
    for (int t = 0; t < 2; ++t)
#pragma unroll
      for (int c = 0; c < 4; ++c) bs[t][c] = (v8f){0,0,0,0,0,0,0,0};
#pragma unroll
    for (int kk = 0; kk < 4; ++kk) {
      v16bf a0 = frag_a(sQC + (rb * 32 + 0) * 128 + kk * 32, lane, 128);
      v16bf a1 = frag_a(sQC + (rb * 32 + 16) * 128 + kk * 32, lane, 128);
#pragma unroll
      for (int c = 0; c < 4; ++c) {
        v16bf bfr = frag_b(sKCT + (kk * 32) * 128, lane, 128, cg * 64 + c * 16);
        bs[0][c] = wmma_bf16(a0, bfr, bs[0][c]);
        bs[1][c] = wmma_bf16(a1, bfr, bs[1][c]);
      }
    }
#pragma unroll
    for (int t = 0; t < 2; ++t)
#pragma unroll
      for (int c = 0; c < 4; ++c)
#pragma unroll
        for (int r = 0; r < 8; ++r) {
          int row = rb * 32 + t * 16 + hf * 8 + r;
          int col = cg * 64 + c * 16 + ln;
          sP[row * 128 + col] = (__bf16)bs[t][c][r];
        }
    __syncthreads();

    // boundary row i0-1 (plain dot products, tiny cost)
    if (i0 > 0 && tid < 128) {
      float s = 0.0f;
      for (int d = 0; d < 128; ++d)
        s += (float)sQm1[d] * (float)sKCT[d * 128 + tid];
      sProw[tid] = s;
    }
    __syncthreads();

    // L1 diff-norm over this bias tile
#pragma unroll 4
    for (int i = 0; i < 64; ++i) {
      int el = tid + i * 256;
      int r = el >> 7, c = el & 127;
      float v = (float)sP[r * 128 + c];
      if (c > 0)       norm += fabsf(v - (float)sP[r * 128 + c - 1]);
      else if (j0 > 0) norm += fabsf(v - sPrevCol[r]);
      if (r > 0)       norm += fabsf(v - (float)sP[(r - 1) * 128 + c]);
      else if (i0 > 0) norm += fabsf(v - sProw[c]);
    }
    __syncthreads();
    if (tid < 128) sPrevCol[tid] = (float)sP[tid * 128 + 127];

    // accb += bias @ V
#pragma unroll
    for (int kk = 0; kk < 4; ++kk) {
      v16bf a0 = frag_a(sP + (rb * 32 + 0) * 128 + kk * 32, lane, 128);
      v16bf a1 = frag_a(sP + (rb * 32 + 16) * 128 + kk * 32, lane, 128);
#pragma unroll
      for (int c = 0; c < 4; ++c) {
        v16bf bfr = frag_b(sV + (kk * 32) * 128, lane, 128, cg * 64 + c * 16);
        accb[0][c] = wmma_bf16(a0, bfr, accb[0][c]);
        accb[1][c] = wmma_bf16(a1, bfr, accb[1][c]);
      }
    }
    __syncthreads();
  }

#pragma unroll
  for (int t = 0; t < 2; ++t)
#pragma unroll
    for (int c = 0; c < 4; ++c)
#pragma unroll
      for (int r = 0; r < 8; ++r) {
        int row = i0 + rb * 32 + t * 16 + hf * 8 + r;
        int col = cg * 64 + c * 16 + ln;
        size_t idx = ((size_t)b * L_ + row) * DQ_ + col;
        out[idx] += accb[t][c][r];
      }
  atomicAdd(normp, norm);
}

// ------------------------------------------------------------------ launch
extern "C" void kernel_launch(void* const* d_in, const int* in_sizes, int n_in,
                              void* d_out, int out_size, void* d_ws, size_t ws_size,
                              hipStream_t stream) {
  (void)in_sizes; (void)n_in; (void)out_size; (void)ws_size;
  const float* query = (const float*)d_in[0];
  const float* key_  = (const float*)d_in[1];
  const float* value = (const float*)d_in[2];
  const float* sf    = (const float*)d_in[3];
  const float* pos   = (const float*)d_in[4];
  const float* Wq  = (const float*)d_in[5];  const float* bq  = (const float*)d_in[6];
  const float* Wk  = (const float*)d_in[7];  const float* bk  = (const float*)d_in[8];
  const float* Wv  = (const float*)d_in[9];  const float* bv  = (const float*)d_in[10];
  const float* W1  = (const float*)d_in[11]; const float* b1  = (const float*)d_in[12];
  const float* W2  = (const float*)d_in[13]; const float* b2  = (const float*)d_in[14];
  const float* Wqc = (const float*)d_in[15]; const float* bqc = (const float*)d_in[16];
  const float* Wkc = (const float*)d_in[17]; const float* bkc = (const float*)d_in[18];

  // Workspace layout (~58 MiB total)
  const size_t NBLD = (size_t)B_ * L_ * DQ_;  // 4,194,304
  __bf16* qb   = (__bf16*)d_ws;
  __bf16* kb   = qb   + NBLD;
  __bf16* vb   = kb   + NBLD;
  __bf16* qcb  = vb   + NBLD;
  __bf16* kcb  = qcb  + NBLD;
  __bf16* wqb  = kcb  + NBLD;
  __bf16* wkb  = wqb  + (size_t)DIN_ * DQ_;
  __bf16* wvb  = wkb  + (size_t)DIN_ * DQ_;
  __bf16* w2b  = wvb  + (size_t)DIN_ * DQ_;
  __bf16* wqcb = w2b  + (size_t)HID_ * DQ_;
  __bf16* wkcb = wqcb + (size_t)DQ_ * DQ_;
  float*  cpe  = (float*)(wkcb + (size_t)DQ_ * DQ_);
  float*  asf  = cpe + NBLD;
  float*  ppos = asf + (size_t)B_ * HID_;

  float* out  = (float*)d_out;
  float* norm = out + NBLD;

  const int NW = DIN_ * DQ_;    // 131072
  const int NS = HID_ * DQ_;    // 16384
  kcvt<<<(NW + 255) / 256, 256, 0, stream>>>(Wq,  wqb,  NW);
  kcvt<<<(NW + 255) / 256, 256, 0, stream>>>(Wk,  wkb,  NW);
  kcvt<<<(NW + 255) / 256, 256, 0, stream>>>(Wv,  wvb,  NW);
  kcvt<<<(NS + 255) / 256, 256, 0, stream>>>(W2,  w2b,  NS);
  kcvt<<<(NS + 255) / 256, 256, 0, stream>>>(Wqc, wqcb, NS);
  kcvt<<<(NS + 255) / 256, 256, 0, stream>>>(Wkc, wkcb, NS);

  kprep<<<((B_ + L_) * HID_ + 255) / 256, 256, 0, stream>>>(sf, pos, W1, b1, asf, ppos);
  knzero<<<1, 32, 0, stream>>>(norm);

  const int MB = (B_ * L_) / 128;  // 256 row-tile blocks
  // q/k/v projections: [32768,1024] @ [1024,128]
  kgemm<<<MB, 256, 0, stream>>>(query, wqb, bq, qb, DIN_, DIN_, 0, 1, nullptr, nullptr);
  kgemm<<<MB, 256, 0, stream>>>(key_,  wkb, bk, kb, DIN_, DIN_, 0, 1, nullptr, nullptr);
  kgemm<<<MB, 256, 0, stream>>>(value, wvb, bv, vb, DIN_, DIN_, 0, 1, nullptr, nullptr);
  // cpe = relu(Asf[b]+Ppos[l]) @ W2 + b2  (fp32 out)
  kgemm<<<MB, 256, 0, stream>>>(nullptr, w2b, b2, cpe, HID_, 0, 1, 0, asf, ppos);
  // qc/kc: [32768,128] @ [128,128]
  kgemm<<<MB, 256, 0, stream>>>(cpe, wqcb, bqc, qcb, HID_, HID_, 0, 1, nullptr, nullptr);
  kgemm<<<MB, 256, 0, stream>>>(cpe, wkcb, bkc, kcb, HID_, HID_, 0, 1, nullptr, nullptr);

  dim3 grid(L_ / 128, B_);
  kflash<<<grid, 256, 0, stream>>>(qb, kb, vb, out);
  kbias<<<grid, 256, 0, stream>>>(qcb, kcb, vb, out, norm);
}
